// GPT2Attention_87522843558459
// MI455X (gfx1250) — matrix operand order repaired and checked by reading, not compiled, not run on hardware
//
#include <hip/hip_runtime.h>
#include <hip/hip_bf16.h>

#define BB 4
#define SS 2048
#define DD 1024
#define HH 16
#define HD 64

typedef __attribute__((ext_vector_type(16))) __bf16       v16bf;
typedef __attribute__((ext_vector_type(8)))  float        v8f;
typedef __attribute__((ext_vector_type(4)))  unsigned int u32x4;

union FragU { v16bf v; u32x4 q[2]; };

__device__ __forceinline__ unsigned short f2bf(float f) {
  unsigned int u = __float_as_uint(f);
  u += 0x7FFFu + ((u >> 16) & 1u);   // round-to-nearest-even
  return (unsigned short)(u >> 16);
}

// ---------------- elementwise f32 -> bf16 ----------------
__global__ void cvt_f32_to_bf16(const float* __restrict__ in,
                                unsigned short* __restrict__ out, int n) {
  int i = blockIdx.x * blockDim.x + threadIdx.x;
  if (i < n) out[i] = f2bf(in[i]);
}

// ---------------- LDS-tiled transpose + convert: W[K,N] f32 -> Wt[N,K] bf16 ----------------
__global__ void transpose_cvt(const float* __restrict__ W,
                              unsigned short* __restrict__ Wt, int K, int N) {
  __shared__ float tile[32][33];
  int n0 = blockIdx.x * 32, k0 = blockIdx.y * 32;
  int tx = threadIdx.x, ty = threadIdx.y;  // 32 x 8
#pragma unroll
  for (int i = 0; i < 32; i += 8)
    tile[ty + i][tx] = W[(size_t)(k0 + ty + i) * N + (n0 + tx)];
  __syncthreads();
#pragma unroll
  for (int i = 0; i < 32; i += 8)
    Wt[(size_t)(n0 + ty + i) * K + (k0 + tx)] = f2bf(tile[tx][ty + i]);
}

// ---------------- split qkv f32 [B*S,3D] -> Q,K [BH,S,64] bf16 ; V^T [BH,64,S] bf16 ----------------
__global__ void split_qkv(const float* __restrict__ qkv,
                          unsigned short* __restrict__ Qb,
                          unsigned short* __restrict__ Kb,
                          unsigned short* __restrict__ Vt, int n) {
  int idx = blockIdx.x * blockDim.x + threadIdx.x;  // over B*S*D
  if (idx >= n) return;
  int bs = idx / DD;
  int c  = idx % DD;
  int h = c / HD, d = c % HD;
  int b = bs / SS, s = bs % SS;
  int bh = b * HH + h;
  const float* row = qkv + (size_t)bs * (3 * DD);
  Qb[((size_t)bh * SS + s) * HD + d] = f2bf(row[c]);
  Kb[((size_t)bh * SS + s) * HD + d] = f2bf(row[DD + c]);
  Vt[((size_t)bh * HD + d) * SS + s] = f2bf(row[2 * DD + c]);
}

// ---------------- bf16 WMMA GEMM: C[M,N] = A[M,K] @ Bt[N,K]^T + bias[N], f32 out ----------------
// One wave computes a 32x64 tile (2x4 grid of 16x16 WMMA tiles), K-step 32.
__global__ void __launch_bounds__(256)
gemm_bf16_wmma(const unsigned short* __restrict__ A,
               const unsigned short* __restrict__ Bt,
               const float* __restrict__ bias,
               float* __restrict__ C, int M, int N, int K) {
  const int lane = threadIdx.x & 31;
  const int wv   = threadIdx.x >> 5;
  const int gw   = blockIdx.x * 8 + wv;
  const int tiles_n = N >> 6;
  const int ntiles  = (M >> 5) * tiles_n;
  if (gw >= ntiles) return;                 // wave-uniform: EXEC stays all-ones
  const int tm = (gw / tiles_n) << 5;
  const int tn = (gw % tiles_n) << 6;
  const int ln = lane & 15, hw = lane >> 4;

  v8f acc[2][4] = {};

#pragma unroll 1   // keep ONE iteration's fragments live: no acc spills to scratch
  for (int k0 = 0; k0 < K; k0 += 32) {
    FragU afr[2], bfr[4];
#pragma unroll
    for (int i = 0; i < 2; ++i) {
      const unsigned short* pa = A + (size_t)(tm + i * 16 + ln) * K + k0 + hw * 8;
      afr[i].q[0] = *(const u32x4*)pa;
      afr[i].q[1] = *(const u32x4*)(pa + 16);
      if (k0 + 64 <= K) __builtin_prefetch(pa + 64, 0, 3);   // WGP-scope prefetch
    }
#pragma unroll
    for (int j = 0; j < 4; ++j) {
      const unsigned short* pb = Bt + (size_t)(tn + j * 16 + ln) * K + k0 + hw * 8;
      bfr[j].q[0] = *(const u32x4*)pb;
      bfr[j].q[1] = *(const u32x4*)(pb + 16);
    }
#pragma unroll
    for (int i = 0; i < 2; ++i)
#pragma unroll
      for (int j = 0; j < 4; ++j)
        acc[i][j] = __builtin_amdgcn_wmma_f32_16x16x32_bf16(
            false, afr[i].v, false, bfr[j].v, (short)0, acc[i][j], false, false);
  }

#pragma unroll
  for (int j = 0; j < 4; ++j) {
    float bv = bias[tn + j * 16 + ln];
#pragma unroll
    for (int i = 0; i < 2; ++i)
#pragma unroll
      for (int r = 0; r < 8; ++r)
        C[(size_t)(tm + i * 16 + r + 8 * hw) * N + (tn + j * 16 + ln)] =
            acc[i][j][r] + bv;
  }
}

// ---------------- flash attention: one wave per 16-row Q tile ----------------
__global__ void __launch_bounds__(128)
attn_wmma(const unsigned short* __restrict__ Q,
          const unsigned short* __restrict__ Kk,
          const unsigned short* __restrict__ Vt,
          unsigned short* __restrict__ ctx) {
  __shared__ __align__(16) unsigned short pshr[4][16][40];  // P bounce, 80B row stride (16-aligned)
  const int lane = threadIdx.x & 31;
  const int wv   = threadIdx.x >> 5;
  const int gw   = blockIdx.x * 4 + wv;
  const int qtiles = SS / 16;
  if (gw >= BB * HH * qtiles) return;       // wave-uniform
  const int bh = gw / qtiles;
  const int q0 = (gw % qtiles) * 16;
  const int b = bh / HH, h = bh % HH;
  const int ln = lane & 15, hw = lane >> 4;

  const unsigned short* Qh = Q  + (size_t)bh * SS * HD;
  const unsigned short* Kh = Kk + (size_t)bh * SS * HD;
  const unsigned short* Vh = Vt + (size_t)bh * HD * SS;

  // Q A-fragments for the two 32-deep d chunks (HD = 64)
  FragU qa[2];
#pragma unroll
  for (int c = 0; c < 2; ++c) {
    const unsigned short* p = Qh + (size_t)(q0 + ln) * HD + c * 32 + hw * 8;
    qa[c].q[0] = *(const u32x4*)p;
    qa[c].q[1] = *(const u32x4*)(p + 16);
  }

  v8f o[4] = {};
  float mrow[8], lrow[8];
#pragma unroll
  for (int r = 0; r < 8; ++r) { mrow[r] = -3.0e38f; lrow[r] = 0.f; }

#pragma unroll 1   // no pipelining across KV chunks: keep register pressure bounded
  for (int kt = 0; kt < q0 + 16; kt += 32) {
    // --- scores: two 16x16 tiles (keys kt..+15, kt+16..+31), each = 2 WMMA over d ---
    v8f s[2] = {};
#pragma unroll
    for (int st = 0; st < 2; ++st) {
      const int keyb = kt + st * 16;
#pragma unroll
      for (int c = 0; c < 2; ++c) {
        FragU kb;
        const unsigned short* p = Kh + (size_t)(keyb + ln) * HD + c * 32 + hw * 8;
        kb.q[0] = *(const u32x4*)p;
        kb.q[1] = *(const u32x4*)(p + 16);
        s[st] = __builtin_amdgcn_wmma_f32_16x16x32_bf16(
            false, qa[c].v, false, kb.v, (short)0, s[st], false, false);
      }
    }
    // --- scale, causal mask, online softmax (rows live in 16-lane groups) ---
#pragma unroll
    for (int r = 0; r < 8; ++r) {
      const int qrow = q0 + r + 8 * hw;
      float s0 = s[0][r] * 0.125f;       // 1/sqrt(64)
      float s1 = s[1][r] * 0.125f;
      if (kt + ln      > qrow) s0 = -3.0e38f;
      if (kt + 16 + ln > qrow) s1 = -3.0e38f;
      float mx = fmaxf(s0, s1);
      mx = fmaxf(mx, __shfl_xor(mx, 1, 32));
      mx = fmaxf(mx, __shfl_xor(mx, 2, 32));
      mx = fmaxf(mx, __shfl_xor(mx, 4, 32));
      mx = fmaxf(mx, __shfl_xor(mx, 8, 32));
      float mnew  = fmaxf(mrow[r], mx);
      float alpha = __expf(mrow[r] - mnew);
      float p0 = __expf(s0 - mnew);
      float p1 = __expf(s1 - mnew);
      float rs = p0 + p1;
      rs += __shfl_xor(rs, 1, 32);
      rs += __shfl_xor(rs, 2, 32);
      rs += __shfl_xor(rs, 4, 32);
      rs += __shfl_xor(rs, 8, 32);
      lrow[r] = lrow[r] * alpha + rs;
      mrow[r] = mnew;
#pragma unroll
      for (int n = 0; n < 4; ++n) o[n][r] *= alpha;
      const int row = r + 8 * hw;
      pshr[wv][row][ln]      = f2bf(p0);   // C-layout -> LDS
      pshr[wv][row][16 + ln] = f2bf(p1);
    }
    asm volatile("s_wait_dscnt 0x0" ::: "memory");  // wave-private LDS re-layout fence
    // --- P as A-fragment (16x32 over keys) ---
    FragU pa;
    pa.q[0] = *(const u32x4*)&pshr[wv][ln][hw * 8];
    pa.q[1] = *(const u32x4*)&pshr[wv][ln][16 + hw * 8];
    // --- o += P @ V : V^T gives key-contiguous B-fragments ---
#pragma unroll
    for (int n = 0; n < 4; ++n) {
      FragU vb;
      const unsigned short* p = Vh + (size_t)(n * 16 + ln) * SS + kt + hw * 8;
      vb.q[0] = *(const u32x4*)p;
      vb.q[1] = *(const u32x4*)(p + 16);
      o[n] = __builtin_amdgcn_wmma_f32_16x16x32_bf16(
          false, pa.v, false, vb.v, (short)0, o[n], false, false);
    }
  }

  // --- normalize and write merged-head ctx [B*S, D] as bf16 ---
#pragma unroll
  for (int r = 0; r < 8; ++r) {
    const float inv = 1.0f / lrow[r];
    const int row = q0 + r + 8 * hw;
    const size_t base = ((size_t)b * SS + row) * DD + (size_t)h * HD;
#pragma unroll
    for (int n = 0; n < 4; ++n)
      ctx[base + n * 16 + ln] = f2bf(o[n][r] * inv);
  }
}

extern "C" void kernel_launch(void* const* d_in, const int* in_sizes, int n_in,
                              void* d_out, int out_size, void* d_ws, size_t ws_size,
                              hipStream_t stream) {
  const float* hidden = (const float*)d_in[0];
  const float* w_attn = (const float*)d_in[1];
  const float* b_attn = (const float*)d_in[2];
  const float* w_proj = (const float*)d_in[3];
  const float* b_proj = (const float*)d_in[4];
  float* out = (float*)d_out;

  size_t off = 0;
  auto carve = [&](size_t bytes) -> void* {
    void* p = (char*)d_ws + off;
    off += (bytes + 255) & ~(size_t)255;
    return p;
  };
  const size_t MS = (size_t)BB * SS;  // 8192 rows
  unsigned short* hid_bf = (unsigned short*)carve(MS * DD * 2);
  unsigned short* wattnT = (unsigned short*)carve((size_t)3 * DD * DD * 2);
  unsigned short* wprojT = (unsigned short*)carve((size_t)DD * DD * 2);
  float*          qkv    = (float*)carve(MS * 3 * DD * 4);
  unsigned short* Qb     = (unsigned short*)carve((size_t)BB * HH * SS * HD * 2);
  unsigned short* Kb     = (unsigned short*)carve((size_t)BB * HH * SS * HD * 2);
  unsigned short* Vt     = (unsigned short*)carve((size_t)BB * HH * SS * HD * 2);
  unsigned short* ctx    = (unsigned short*)carve(MS * DD * 2);

  {
    int n = (int)(MS * DD);
    cvt_f32_to_bf16<<<(n + 255) / 256, 256, 0, stream>>>(hidden, hid_bf, n);
  }
  transpose_cvt<<<dim3(3 * DD / 32, DD / 32), dim3(32, 8), 0, stream>>>(w_attn, wattnT, DD, 3 * DD);
  transpose_cvt<<<dim3(DD / 32, DD / 32), dim3(32, 8), 0, stream>>>(w_proj, wprojT, DD, DD);

  {  // qkv = hidden @ w_attn + b_attn
    int M = (int)MS, N = 3 * DD, K = DD;
    int waves = (M / 32) * (N / 64);
    gemm_bf16_wmma<<<(waves + 7) / 8, 256, 0, stream>>>(hid_bf, wattnT, b_attn, qkv, M, N, K);
  }
  {
    int n = (int)(MS * DD);
    split_qkv<<<(n + 255) / 256, 256, 0, stream>>>(qkv, Qb, Kb, Vt, n);
  }
  {
    int waves = BB * HH * (SS / 16);
    attn_wmma<<<(waves + 3) / 4, 128, 0, stream>>>(Qb, Kb, Vt, ctx);
  }
  {  // out = ctx @ w_proj + b_proj
    int M = (int)MS, N = DD, K = DD;
    int waves = (M / 32) * (N / 64);
    gemm_bf16_wmma<<<(waves + 7) / 8, 256, 0, stream>>>(ctx, wprojT, b_proj, out, M, N, K);
  }
  (void)in_sizes; (void)n_in; (void)out_size; (void)ws_size;
}
